// ImageRenderer_48026324304156
// MI455X (gfx1250) — matrix operand order repaired and checked by reading, not compiled, not run on hardware
//
#include <hip/hip_runtime.h>
#include <hip/hip_bf16.h>

typedef __attribute__((ext_vector_type(16))) _Float16 v16h;
typedef __attribute__((ext_vector_type(8)))  _Float16 v8h;
typedef __attribute__((ext_vector_type(8)))  float    v8f;

namespace {
constexpr int   kW     = 256;
constexpr int   kH     = 256;
constexpr int   kR     = kW * kH;     // 65536 rays
constexpr int   kS     = 64;          // samples per ray
constexpr int   kHid   = 64;          // hidden width
constexpr float kZN    = 2.0f;
constexpr float kZF    = 6.0f;
constexpr float kBinDt = (kZF - kZN) / (float)kS;   // 0.0625
}

__device__ __forceinline__ float sigmoidf(float x) {
  return 1.0f / (1.0f + __expf(-x));
}

// One wave (32 lanes) renders one ray.
// Layer 1:  A[16x32] (6 real K)  x  B[32x16] (W1 tile)  -> C[16x16], 4 N-tiles
// Layer 2:  A[16x32] (h chunk)   x  B[32x16] (W2 pad )  -> C[16x16], K-chained x2
// Composite: wave-parallel prefix-product scan (2 samples per lane).
__global__ __launch_bounds__(256) void nerf_render_kernel(
    const float* __restrict__ cam,   // 4x4 camera_local_to_world
    const float* __restrict__ u,     // R x S jitter
    const float* __restrict__ crd,   // R x 3 cam-space ray dirs
    const float* __restrict__ W1,    // 6 x 64
    const float* __restrict__ b1,    // 64
    const float* __restrict__ W2,    // 64 x 4
    const float* __restrict__ b2,    // 4
    float* __restrict__ outc)        // R x 3
{
  __shared__ __align__(16) _Float16 sh_h[8][16 * kHid]; // per-wave h (layer1->layer2 transpose)
  __shared__ __align__(16) float sh_o[8][kS * 4];        // per-wave raw MLP outputs
  __shared__ float sh_t[8][kS];                          // per-wave sample depths

  const int lane   = threadIdx.x & 31;
  const int col    = lane & 15;     // C-layout column / A-layout row
  const int halfId = lane >> 4;
  const int wv     = threadIdx.x >> 5;
  const int ray    = blockIdx.x * 8 + wv;

  __builtin_prefetch(u + (size_t)ray * kS, 0, 0);

  // ---------------- weights into WMMA B layouts (once per wave) ----------------
  // B layout (16-bit, 32x16): lane holds column n'=lane&15; element e -> K = (e<8 ? e : e+8) + 8*halfId.
  v16h B1[4];
#pragma unroll
  for (int j = 0; j < 4; ++j) {
    v16h b = {};
    if (halfId == 0) {
#pragma unroll
      for (int e = 0; e < 6; ++e)            // real K = 0..5 only
        b[e] = (_Float16)W1[e * kHid + 16 * j + col];
    }
    B1[j] = b;
  }
  v16h B2m[2];
#pragma unroll
  for (int kk = 0; kk < 2; ++kk) {
    v16h b = {};
    if (col < 4) {                            // only 4 real output columns
#pragma unroll
      for (int e = 0; e < 16; ++e) {
        const int K = 32 * kk + (e < 8 ? e : e + 8) + 8 * halfId;
        b[e] = (_Float16)W2[K * 4 + col];
      }
    }
    B2m[kk] = b;
  }
  float b1v[4];
#pragma unroll
  for (int j = 0; j < 4; ++j) b1v[j] = b1[16 * j + col];
  const float b2v = (col < 4) ? b2[col] : 0.0f;

  // ---------------- per-ray geometry ----------------
  const float cdx = crd[(size_t)ray * 3 + 0];
  const float cdy = crd[(size_t)ray * 3 + 1];
  const float cdz = crd[(size_t)ray * 3 + 2];
  const float dx = cam[0] * cdx + cam[1] * cdy + cam[2]  * cdz;
  const float dy = cam[4] * cdx + cam[5] * cdy + cam[6]  * cdz;
  const float dz = cam[8] * cdx + cam[9] * cdy + cam[10] * cdz;
  const float ox = cam[3], oy = cam[7], oz = cam[11];

  // ---------------- 4 M-tiles of 16 samples ----------------
#pragma unroll
  for (int tile = 0; tile < 4; ++tile) {
    const int   s  = tile * 16 + col;
    const float uu = u[(size_t)ray * kS + s];
    const float ts = kZN + ((float)s + uu) * kBinDt;
    if (halfId == 0) sh_t[wv][s] = ts;

    // A layout (16-bit, 16x32): lane holds row m=lane&15;
    // element e -> K = (e<8 ? e : e+8) + 8*halfId => features K=0..5 live in halfId==0, e=0..5.
    const float feat[6] = { ox + dx * ts, oy + dy * ts, oz + dz * ts, dx, dy, dz };
    v16h A = {};
    if (halfId == 0) {
#pragma unroll
      for (int e = 0; e < 6; ++e) A[e] = (_Float16)feat[e];
    }

    // ---- layer 1: h = relu(x @ W1 + b1) -> LDS (f16) ----
#pragma unroll
    for (int j = 0; j < 4; ++j) {
      v8f c;
#pragma unroll
      for (int v = 0; v < 8; ++v) c[v] = b1v[j];
      c = __builtin_amdgcn_wmma_f32_16x16x32_f16(false, A, false, B1[j],
                                                 (short)0, c, false, false);
      // C layout: lane column n = 16j+col, VGPR v -> row = v + 8*halfId
#pragma unroll
      for (int v = 0; v < 8; ++v) {
        const int row = v + 8 * halfId;
        const float hv = c[v] > 0.0f ? c[v] : 0.0f;
        sh_h[wv][row * kHid + 16 * j + col] = (_Float16)hv;
      }
    }

    // ---- layer 2: out = h @ W2 + b2 (two chained K=32 chunks) ----
    v8f c2;
#pragma unroll
    for (int v = 0; v < 8; ++v) c2[v] = b2v;
#pragma unroll
    for (int kk = 0; kk < 2; ++kk) {
      const int base = 32 * kk;
      // A-layout reads: two contiguous 16B runs per lane -> ds_load_b128
      const v8h lo = *(const v8h*)&sh_h[wv][col * kHid + base + 8 * halfId];
      const v8h hi = *(const v8h*)&sh_h[wv][col * kHid + base + 16 + 8 * halfId];
      const v16h Ah = __builtin_shufflevector(lo, hi,
          0, 1, 2, 3, 4, 5, 6, 7, 8, 9, 10, 11, 12, 13, 14, 15);
      c2 = __builtin_amdgcn_wmma_f32_16x16x32_f16(false, Ah, false, B2m[kk],
                                                  (short)0, c2, false, false);
    }

    if (col < 4) {
#pragma unroll
      for (int v = 0; v < 8; ++v) {
        const int row = v + 8 * halfId;
        sh_o[wv][(tile * 16 + row) * 4 + col] = c2[v];
      }
    }
  }

  // ---------------- wave-parallel alpha compositing ----------------
  // Each lane owns samples s0 = 2*lane, s1 = 2*lane+1.
  // q[s] = 1 - a[s] = exp(-relu(d[s]) * sep[s]); T is an exclusive prefix product of q.
  {
    const int s0 = 2 * lane;
    const int s1 = s0 + 1;

    const float4 o0 = *(const float4*)&sh_o[wv][s0 * 4];  // ds_load_b128
    const float4 o1 = *(const float4*)&sh_o[wv][s1 * 4];

    const float t0 = sh_t[wv][s0];
    const float t1 = sh_t[wv][s1];
    const float t2 = (s1 < kS - 1) ? sh_t[wv][s1 + 1] : 0.0f;

    const float d0 = o0.x > 0.0f ? o0.x : 0.0f;
    const float d1 = o1.x > 0.0f ? o1.x : 0.0f;
    const float sep0 = t1 - t0;
    const float sep1 = (s1 < kS - 1) ? (t2 - t1) : kBinDt;

    const float q0 = __expf(-d0 * sep0);
    const float q1 = __expf(-d1 * sep1);
    const float a0 = 1.0f - q0;
    const float a1 = 1.0f - q1;

    // Inclusive product scan of p = q0*q1 over the 32 lanes, then make exclusive.
    float incl = q0 * q1;
#pragma unroll
    for (int dd = 1; dd < 32; dd <<= 1) {
      const float up = __shfl_up(incl, dd, 32);
      if (lane >= dd) incl *= up;
    }
    float excl = __shfl_up(incl, 1, 32);
    if (lane == 0) excl = 1.0f;

    const float T0 = excl;        // transmittance before s0
    const float T1 = excl * q0;   // transmittance before s1
    const float w0 = a0 * T0;
    const float w1 = a1 * T1;

    float cr = w0 * sigmoidf(o0.y) + w1 * sigmoidf(o1.y);
    float cg = w0 * sigmoidf(o0.z) + w1 * sigmoidf(o1.z);
    float cb = w0 * sigmoidf(o0.w) + w1 * sigmoidf(o1.w);

    // Butterfly sum across the wave.
#pragma unroll
    for (int dd = 16; dd >= 1; dd >>= 1) {
      cr += __shfl_xor(cr, dd, 32);
      cg += __shfl_xor(cg, dd, 32);
      cb += __shfl_xor(cb, dd, 32);
    }

    if (lane < 3) {
      const float val = (lane == 0) ? cr : ((lane == 1) ? cg : cb);
      outc[(size_t)ray * 3 + lane] = val;
    }
  }
}

extern "C" void kernel_launch(void* const* d_in, const int* in_sizes, int n_in,
                              void* d_out, int out_size, void* d_ws, size_t ws_size,
                              hipStream_t stream) {
  const float* cam = (const float*)d_in[0];  // camera_local_to_world (16)
  const float* u   = (const float*)d_in[1];  // u (R*S)
  const float* crd = (const float*)d_in[2];  // cam_ray_dirs (R*3)
  const float* W1  = (const float*)d_in[3];  // 6*64
  const float* b1  = (const float*)d_in[4];  // 64
  const float* W2  = (const float*)d_in[5];  // 64*4
  const float* b2  = (const float*)d_in[6];  // 4
  float* outc = (float*)d_out;               // R*3

  dim3 grid(kR / 8);   // 8 rays (waves) per 256-thread block
  dim3 block(256);
  hipLaunchKernelGGL(nerf_render_kernel, grid, block, 0, stream,
                     cam, u, crd, W1, b1, W2, b2, outc);
}
